// BasicBlock_deformable_6433861009839
// MI455X (gfx1250) — matrix operand order repaired and checked by reading, not compiled, not run on hardware
//
#include <hip/hip_runtime.h>
#include <hip/hip_bf16.h>

// ---------------- types ----------------
typedef __bf16 bf16_t;
typedef __attribute__((ext_vector_type(16))) __bf16 v16bf;
typedef __attribute__((ext_vector_type(8)))  __bf16 v8bf;
typedef __attribute__((ext_vector_type(8)))  float  v8f;

#define B_    16
#define C_    128
#define H_    56
#define W_    56
#define HW_   3136          // 56*56
#define NHW_  50176         // B*HW
#define K_    1152          // C*9
#define KSTEPS 36           // 1152/32
#define TILE_N 128
#define NBLK  392           // 50176/128

// Async copy of 32 bytes global->LDS (two b128), tracked by ASYNCcnt.
__device__ __forceinline__ void async_copy32(uint32_t ldsOff, const void* g) {
  asm volatile(
      "global_load_async_to_lds_b128 %0, %1, off\n\t"
      "global_load_async_to_lds_b128 %0, %1, off offset:16"
      :: "v"(ldsOff), "v"(g) : "memory");
}
__device__ __forceinline__ void wait_async() {
  asm volatile("s_wait_asynccnt 0" ::: "memory");
}

// ---------------- conversion kernels ----------------
__global__ __launch_bounds__(256)
void cvt_f32_bf16(const float* __restrict__ in, bf16_t* __restrict__ out, int n) {
  int i = blockIdx.x * 256 + threadIdx.x;
  if (i < n) out[i] = (bf16_t)in[i];
}

// pad w_off (18x1152) into 32x1152 with zero rows
__global__ __launch_bounds__(256)
void cvt_woff(const float* __restrict__ in, bf16_t* __restrict__ out) {
  int i = blockIdx.x * 256 + threadIdx.x;
  if (i < 32 * K_) {
    int row = i / K_;
    out[i] = (row < 18) ? (bf16_t)in[i] : (bf16_t)0.f;
  }
}

// ---------------- conv (3x3, pad 1) implicit GEMM: M=128 K=1152 N=50176 ----
// Block tile: M=128 x N=128, K-step 32. 8 waves: wave w -> M-pair (w&3), N-group (w>>2).
__global__ __launch_bounds__(256)
void conv_gemm_bf16(const bf16_t* __restrict__ Aw,
                    const bf16_t* __restrict__ Xb,
                    float* __restrict__ Y) {
  __shared__ bf16_t Atile[128][32];     // [m][k]
  __shared__ bf16_t Btile[TILE_N][32];  // [n][k]
  __shared__ int sOff[TILE_N][9];       // safe spatial offset per (col, tap)
  __shared__ int sVal[TILE_N][9];       // validity flag

  const int t    = threadIdx.x;
  const int wave = t >> 5;
  const int lane = t & 31;
  const int half = lane >> 4;
  const int col0 = blockIdx.x * TILE_N;

  // B-build mapping: one column, 16 consecutive k per thread
  const int j     = t & (TILE_N - 1);
  const int khalf = t >> 7;
  const int cg = col0 + j;
  const int b  = cg / HW_;
  const int hw = cg - b * HW_;
  const int h  = hw / W_;
  const int w  = hw - h * W_;
  const size_t xBase = (size_t)b * C_ * HW_ + h * W_ + w;

  // ---- precompute per-column tap tables ----
  for (int it = t; it < TILE_N * 9; it += 256) {
    const int jj  = it / 9;
    const int tap = it - jj * 9;
    const int cgl = col0 + jj;
    const int bb  = cgl / HW_;
    const int hwl = cgl - bb * HW_;
    const int hh  = hwl / W_;
    const int ww  = hwl - hh * W_;
    const int dh = tap / 3 - 1, dw = tap % 3 - 1;
    const int hn = hh + dh, wn = ww + dw;
    const bool ok = (hn >= 0 && hn < H_ && wn >= 0 && wn < W_);
    sOff[jj][tap] = ok ? dh * W_ + dw : 0;
    sVal[jj][tap] = ok ? 1 : 0;
  }

  const int am = t >> 1;     // A-tile row
  const int ah = t & 1;      // 16-element half
  const int mpair = wave & 3;
  const int ngrp  = wave >> 2;

  v8f acc[2][4] = {};

  for (int ks = 0; ks < KSTEPS; ++ks) {
    const int k0 = ks * 32;
    __syncthreads();
    // ---- async stage A tile (32 B per lane, overlapped with B gather) ----
    async_copy32((uint32_t)(uintptr_t)&Atile[am][16 * ah],
                 Aw + am * K_ + k0 + 16 * ah);
    // ---- build B tile: incremental (cin, tap) walk, branchless gather ----
    {
      const int kg0 = k0 + khalf * 16;
      int cin = kg0 / 9;
      int tap = kg0 - cin * 9;
      const bf16_t* p = Xb + xBase + (size_t)cin * HW_;
      #pragma unroll
      for (int i = 0; i < 16; ++i) {
        const int off = sOff[j][tap];
        bf16_t v = p[off];
        Btile[j][khalf * 16 + i] = sVal[j][tap] ? v : (bf16_t)0.f;
        if (++tap == 9) { tap = 0; p += HW_; }
      }
    }
    wait_async();
    __syncthreads();
    // ---- fragments + 8 WMMA ----
    union AU { v16bf v; v8bf hlf[2]; } af[2];
    #pragma unroll
    for (int mi = 0; mi < 2; ++mi) {
      const int arow = (mpair * 2 + mi) * 16 + (lane & 15);
      af[mi].hlf[0] = *reinterpret_cast<const v8bf*>(&Atile[arow][8 * half]);
      af[mi].hlf[1] = *reinterpret_cast<const v8bf*>(&Atile[arow][16 + 8 * half]);
    }
    #pragma unroll
    for (int js = 0; js < 4; ++js) {
      const int bcol = ngrp * 64 + 16 * js + (lane & 15);
      v16bf bfrag = __builtin_shufflevector(*reinterpret_cast<const v8bf*>(&Btile[bcol][8 * half]), *reinterpret_cast<const v8bf*>(&Btile[bcol][16 + 8 * half]), 0, 1, 2, 3, 4, 5, 6, 7, 8, 9, 10, 11, 12, 13, 14, 15);
      #pragma unroll
      for (int mi = 0; mi < 2; ++mi)
        acc[mi][js] = __builtin_amdgcn_wmma_f32_16x16x32_bf16(
            false, af[mi].v, false, bfrag, (short)0, acc[mi][js], false, false);
    }
  }
  // ---- epilogue ----
  const int nn = lane & 15;
  #pragma unroll
  for (int js = 0; js < 4; ++js) {
    const int cgo = col0 + ngrp * 64 + 16 * js + nn;
    const int bo  = cgo / HW_;
    const int hwo = cgo - bo * HW_;
    #pragma unroll
    for (int mi = 0; mi < 2; ++mi) {
      #pragma unroll
      for (int r = 0; r < 8; ++r) {
        const int cout = (mpair * 2 + mi) * 16 + r + 8 * half;
        Y[((size_t)(bo * C_ + cout)) * HW_ + hwo] = acc[mi][js][r];
      }
    }
  }
}

// ---------------- offset conv: M=32(18 valid) K=1152 N=50176 ----------------
// 8 waves: wave w -> M-tile (w&1), N-pair (w>>1) of two 16-col subtiles.
__global__ __launch_bounds__(256)
void off_gemm_bf16(const bf16_t* __restrict__ Aw,      // [32][1152]
                   const bf16_t* __restrict__ Xb,      // o1b
                   const float* __restrict__ boff,     // [18]
                   float* __restrict__ OFF) {          // [B][18][HW]
  __shared__ bf16_t Atile[32][32];
  __shared__ bf16_t Btile[TILE_N][32];
  __shared__ int sOff[TILE_N][9];
  __shared__ int sVal[TILE_N][9];

  const int t    = threadIdx.x;
  const int wave = t >> 5;
  const int lane = t & 31;
  const int half = lane >> 4;
  const int col0 = blockIdx.x * TILE_N;
  const int mt    = wave & 1;
  const int npair = wave >> 1;

  const int j     = t & (TILE_N - 1);
  const int khalf = t >> 7;
  const int cg = col0 + j;
  const int b  = cg / HW_;
  const int hw = cg - b * HW_;
  const int h  = hw / W_;
  const int w  = hw - h * W_;
  const size_t xBase = (size_t)b * C_ * HW_ + h * W_ + w;

  for (int it = t; it < TILE_N * 9; it += 256) {
    const int jj  = it / 9;
    const int tap = it - jj * 9;
    const int cgl = col0 + jj;
    const int bb  = cgl / HW_;
    const int hwl = cgl - bb * HW_;
    const int hh  = hwl / W_;
    const int ww  = hwl - hh * W_;
    const int dh = tap / 3 - 1, dw = tap % 3 - 1;
    const int hn = hh + dh, wn = ww + dw;
    const bool ok = (hn >= 0 && hn < H_ && wn >= 0 && wn < W_);
    sOff[jj][tap] = ok ? dh * W_ + dw : 0;
    sVal[jj][tap] = ok ? 1 : 0;
  }

  v8f acc[2] = {};

  for (int ks = 0; ks < KSTEPS; ++ks) {
    const int k0 = ks * 32;
    __syncthreads();
    if (t < 64) {   // 32 rows x 2 halves
      const int m = t >> 1, hh2 = t & 1;
      async_copy32((uint32_t)(uintptr_t)&Atile[m][16 * hh2],
                   Aw + m * K_ + k0 + 16 * hh2);
    }
    {
      const int kg0 = k0 + khalf * 16;
      int cin = kg0 / 9;
      int tap = kg0 - cin * 9;
      const bf16_t* p = Xb + xBase + (size_t)cin * HW_;
      #pragma unroll
      for (int i = 0; i < 16; ++i) {
        const int off = sOff[j][tap];
        bf16_t v = p[off];
        Btile[j][khalf * 16 + i] = sVal[j][tap] ? v : (bf16_t)0.f;
        if (++tap == 9) { tap = 0; p += HW_; }
      }
    }
    wait_async();
    __syncthreads();
    const int arow = mt * 16 + (lane & 15);
    union { v16bf v; v8bf hlf[2]; } afrag;
    afrag.hlf[0] = *reinterpret_cast<const v8bf*>(&Atile[arow][8 * half]);
    afrag.hlf[1] = *reinterpret_cast<const v8bf*>(&Atile[arow][16 + 8 * half]);
    #pragma unroll
    for (int q = 0; q < 2; ++q) {
      const int bcol = (npair * 2 + q) * 16 + (lane & 15);
      v16bf bfrag = __builtin_shufflevector(*reinterpret_cast<const v8bf*>(&Btile[bcol][8 * half]), *reinterpret_cast<const v8bf*>(&Btile[bcol][16 + 8 * half]), 0, 1, 2, 3, 4, 5, 6, 7, 8, 9, 10, 11, 12, 13, 14, 15);
      acc[q] = __builtin_amdgcn_wmma_f32_16x16x32_bf16(
          false, afrag.v, false, bfrag, (short)0, acc[q], false, false);
    }
  }
  const int nn = lane & 15;
  #pragma unroll
  for (int q = 0; q < 2; ++q) {
    const int cgo = col0 + (npair * 2 + q) * 16 + nn;
    const int bo  = cgo / HW_;
    const int hwo = cgo - bo * HW_;
    #pragma unroll
    for (int r = 0; r < 8; ++r) {
      const int cout = mt * 16 + r + 8 * half;
      if (cout < 18)
        OFF[((size_t)(bo * 18 + cout)) * HW_ + hwo] = acc[q][r] + boff[cout];
    }
  }
}

// ---------------- deform conv: fused bilinear gather + GEMM ----------------
__global__ __launch_bounds__(256)
void deform_gemm_bf16(const bf16_t* __restrict__ Aw,    // w2 bf16 [128][1152]
                      const bf16_t* __restrict__ Xb,    // o1b [B][C][H][W]
                      const float* __restrict__ OFF,    // [B][18][HW]
                      float* __restrict__ Y) {
  __shared__ bf16_t Atile[128][32];
  __shared__ bf16_t Btile[TILE_N][32];
  __shared__ int   sIdx[TILE_N][9][4];
  __shared__ float sWgt[TILE_N][9][4];

  const int t    = threadIdx.x;
  const int wave = t >> 5;
  const int lane = t & 31;
  const int half = lane >> 4;
  const int col0 = blockIdx.x * TILE_N;

  const int j     = t & (TILE_N - 1);
  const int khalf = t >> 7;
  const int cg = col0 + j;
  const int b  = cg / HW_;
  const int hw = cg - b * HW_;
  const size_t xBase = (size_t)b * C_ * HW_;

  // ---- precompute bilinear corners (branchless: invalid -> idx 0, wgt 0) ----
  for (int it = t; it < TILE_N * 9; it += 256) {
    const int jj  = it / 9;
    const int tap = it - jj * 9;
    const int cgl = col0 + jj;
    const int bb  = cgl / HW_;
    const int hwl = cgl - bb * HW_;
    const int hh  = hwl / W_;
    const int ww  = hwl - hh * W_;
    const float o1 = OFF[((size_t)(bb * 18 + tap)) * HW_ + hwl];
    const float o2 = OFF[((size_t)(bb * 18 + 9 + tap)) * HW_ + hwl];
    float px = (float)(hh + 1 + tap / 3 - 1) + o1;
    float py = (float)(ww + 1 + tap % 3 - 1) + o2;
    px = fminf(fmaxf(px, 0.f), (float)(H_ + 1));
    py = fminf(fmaxf(py, 0.f), (float)(W_ + 1));
    const float fx = floorf(px), fy = floorf(py);
    const float x0 = fminf(fmaxf(fx, 0.f), (float)(H_ + 1));
    const float x1 = fminf(fmaxf(fx + 1.f, 0.f), (float)(H_ + 1));
    const float y0 = fminf(fmaxf(fy, 0.f), (float)(W_ + 1));
    const float y1 = fminf(fmaxf(fy + 1.f, 0.f), (float)(W_ + 1));
    const float glt = (1.f + (x0 - px)) * (1.f + (y0 - py));
    const float grb = (1.f - (x1 - px)) * (1.f - (y1 - py));
    const float glb = (1.f + (x0 - px)) * (1.f - (y1 - py));
    const float grt = (1.f - (x1 - px)) * (1.f + (y0 - py));
    const int ix0 = (int)x0, ix1 = (int)x1, iy0 = (int)y0, iy1 = (int)y1;
    auto emit = [&](int c, int ix, int iy, float g) {
      const bool ok = (ix >= 1 && ix <= H_ && iy >= 1 && iy <= W_);
      sIdx[jj][tap][c] = ok ? (ix - 1) * W_ + (iy - 1) : 0;
      sWgt[jj][tap][c] = ok ? g : 0.f;
    };
    emit(0, ix0, iy0, glt);
    emit(1, ix1, iy1, grb);
    emit(2, ix0, iy1, glb);
    emit(3, ix1, iy0, grt);
  }

  const int am = t >> 1;
  const int ah = t & 1;
  const int mpair = wave & 3;
  const int ngrp  = wave >> 2;
  v8f acc[2][4] = {};

  for (int ks = 0; ks < KSTEPS; ++ks) {
    const int k0 = ks * 32;
    __syncthreads();
    async_copy32((uint32_t)(uintptr_t)&Atile[am][16 * ah],
                 Aw + am * K_ + k0 + 16 * ah);
    {
      const int kg0 = k0 + khalf * 16;
      int cin = kg0 / 9;
      int tap = kg0 - cin * 9;
      const bf16_t* p = Xb + xBase + (size_t)cin * HW_;
      #pragma unroll
      for (int i = 0; i < 16; ++i) {
        const int*   ip = &sIdx[j][tap][0];
        const float* wp = &sWgt[j][tap][0];
        float v = wp[0] * (float)p[ip[0]] + wp[1] * (float)p[ip[1]]
                + wp[2] * (float)p[ip[2]] + wp[3] * (float)p[ip[3]];
        Btile[j][khalf * 16 + i] = (bf16_t)v;
        if (++tap == 9) { tap = 0; p += HW_; }
      }
    }
    wait_async();
    __syncthreads();
    union AU { v16bf v; v8bf hlf[2]; } af[2];
    #pragma unroll
    for (int mi = 0; mi < 2; ++mi) {
      const int arow = (mpair * 2 + mi) * 16 + (lane & 15);
      af[mi].hlf[0] = *reinterpret_cast<const v8bf*>(&Atile[arow][8 * half]);
      af[mi].hlf[1] = *reinterpret_cast<const v8bf*>(&Atile[arow][16 + 8 * half]);
    }
    #pragma unroll
    for (int js = 0; js < 4; ++js) {
      const int bcol = ngrp * 64 + 16 * js + (lane & 15);
      v16bf bfrag = __builtin_shufflevector(*reinterpret_cast<const v8bf*>(&Btile[bcol][8 * half]), *reinterpret_cast<const v8bf*>(&Btile[bcol][16 + 8 * half]), 0, 1, 2, 3, 4, 5, 6, 7, 8, 9, 10, 11, 12, 13, 14, 15);
      #pragma unroll
      for (int mi = 0; mi < 2; ++mi)
        acc[mi][js] = __builtin_amdgcn_wmma_f32_16x16x32_bf16(
            false, af[mi].v, false, bfrag, (short)0, acc[mi][js], false, false);
    }
  }
  const int nn = lane & 15;
  #pragma unroll
  for (int js = 0; js < 4; ++js) {
    const int cgo = col0 + ngrp * 64 + 16 * js + nn;
    const int bo  = cgo / HW_;
    const int hwo = cgo - bo * HW_;
    #pragma unroll
    for (int mi = 0; mi < 2; ++mi) {
      #pragma unroll
      for (int r = 0; r < 8; ++r) {
        const int cout = (mpair * 2 + mi) * 16 + r + 8 * half;
        Y[((size_t)(bo * C_ + cout)) * HW_ + hwo] = acc[mi][js][r];
      }
    }
  }
}

// ---------------- BN statistics (one block per channel) ----------------
__global__ __launch_bounds__(256)
void bn_stats(const float* __restrict__ y, float* __restrict__ mean,
              float* __restrict__ istd) {
  __shared__ float rs[256], rq[256];
  const int c = blockIdx.x;
  float s = 0.f, q = 0.f;
  for (int i = threadIdx.x; i < B_ * HW_; i += 256) {
    const int b  = i / HW_;
    const int hw = i - b * HW_;
    const float v = y[((size_t)(b * C_ + c)) * HW_ + hw];
    s += v; q += v * v;
  }
  rs[threadIdx.x] = s; rq[threadIdx.x] = q;
  __syncthreads();
  for (int off = 128; off > 0; off >>= 1) {
    if (threadIdx.x < off) {
      rs[threadIdx.x] += rs[threadIdx.x + off];
      rq[threadIdx.x] += rq[threadIdx.x + off];
    }
    __syncthreads();
  }
  if (threadIdx.x == 0) {
    const float inv = 1.f / (float)(B_ * HW_);
    const float m = rs[0] * inv;
    const float var = rq[0] * inv - m * m;
    mean[c] = m;
    istd[c] = rsqrtf(var + 1e-5f);
  }
}

// ---------------- BN + ReLU -> bf16 ----------------
__global__ __launch_bounds__(256)
void bn_relu_to_bf16(const float* __restrict__ y, const float* __restrict__ g,
                     const float* __restrict__ bb, const float* __restrict__ mean,
                     const float* __restrict__ istd, bf16_t* __restrict__ out) {
  const int i = blockIdx.x * 256 + threadIdx.x;
  if (i < B_ * C_ * HW_) {
    const int c = (i / HW_) % C_;
    float v = g[c] * (y[i] - mean[c]) * istd[c] + bb[c];
    out[i] = (bf16_t)(v > 0.f ? v : 0.f);
  }
}

// ---------------- final: BN2 + residual + ReLU -> f32 ----------------
__global__ __launch_bounds__(256)
void bn_add_relu(const float* __restrict__ y, const float* __restrict__ x,
                 const float* __restrict__ g, const float* __restrict__ bb,
                 const float* __restrict__ mean, const float* __restrict__ istd,
                 float* __restrict__ out) {
  const int i = blockIdx.x * 256 + threadIdx.x;
  if (i < B_ * C_ * HW_) {
    const int c = (i / HW_) % C_;
    float v = g[c] * (y[i] - mean[c]) * istd[c] + bb[c] + x[i];
    out[i] = v > 0.f ? v : 0.f;
  }
}

// ---------------- launcher ----------------
extern "C" void kernel_launch(void* const* d_in, const int* in_sizes, int n_in,
                              void* d_out, int out_size, void* d_ws, size_t ws_size,
                              hipStream_t stream) {
  const float* x     = (const float*)d_in[0];
  const float* w1    = (const float*)d_in[1];
  const float* g1    = (const float*)d_in[2];
  const float* b1    = (const float*)d_in[3];
  const float* w_off = (const float*)d_in[4];
  const float* b_off = (const float*)d_in[5];
  const float* w2    = (const float*)d_in[6];
  const float* g2    = (const float*)d_in[7];
  const float* b2    = (const float*)d_in[8];
  float* out = (float*)d_out;

  char* ws = (char*)d_ws;
  auto alloc = [&](size_t bytes) {
    char* p = ws;
    ws += (bytes + 255) & ~(size_t)255;
    return (void*)p;
  };
  bf16_t* w1b  = (bf16_t*)alloc((size_t)C_ * K_ * 2);
  bf16_t* w2b  = (bf16_t*)alloc((size_t)C_ * K_ * 2);
  bf16_t* wob  = (bf16_t*)alloc((size_t)32 * K_ * 2);
  bf16_t* xb   = (bf16_t*)alloc((size_t)B_ * C_ * HW_ * 2);
  bf16_t* o1b  = (bf16_t*)alloc((size_t)B_ * C_ * HW_ * 2);
  float*  y1   = (float*)alloc((size_t)B_ * C_ * HW_ * 4);
  float*  y2   = (float*)alloc((size_t)B_ * C_ * HW_ * 4);
  float*  offs = (float*)alloc((size_t)B_ * 18 * HW_ * 4);
  float*  mean1 = (float*)alloc(512);
  float*  istd1 = (float*)alloc(512);
  float*  mean2 = (float*)alloc(512);
  float*  istd2 = (float*)alloc(512);

  const int nElem = B_ * C_ * HW_;          // 6,422,528
  const int eBlks = (nElem + 255) / 256;

  cvt_f32_bf16<<<eBlks, 256, 0, stream>>>(x, xb, nElem);
  cvt_f32_bf16<<<(C_ * K_ + 255) / 256, 256, 0, stream>>>(w1, w1b, C_ * K_);
  cvt_f32_bf16<<<(C_ * K_ + 255) / 256, 256, 0, stream>>>(w2, w2b, C_ * K_);
  cvt_woff<<<(32 * K_ + 255) / 256, 256, 0, stream>>>(w_off, wob);

  conv_gemm_bf16<<<NBLK, 256, 0, stream>>>(w1b, xb, y1);
  bn_stats<<<C_, 256, 0, stream>>>(y1, mean1, istd1);
  bn_relu_to_bf16<<<eBlks, 256, 0, stream>>>(y1, g1, b1, mean1, istd1, o1b);

  off_gemm_bf16<<<NBLK, 256, 0, stream>>>(wob, o1b, b_off, offs);

  deform_gemm_bf16<<<NBLK, 256, 0, stream>>>(w2b, o1b, offs, y2);
  bn_stats<<<C_, 256, 0, stream>>>(y2, mean2, istd2);
  bn_add_relu<<<eBlks, 256, 0, stream>>>(y2, x, g2, b2, mean2, istd2, out);
}
